// TPM_59072980189676
// MI455X (gfx1250) — matrix operand order repaired and checked by reading, not compiled, not run on hardware
//
#include <hip/hip_runtime.h>

#define CDIM 64
#define NDIM 4096
#define NB   4

typedef __bf16 bf16_t;
typedef __attribute__((ext_vector_type(16))) __bf16 v16bf;
typedef __attribute__((ext_vector_type(8)))  __bf16 v8bf;
typedef __attribute__((ext_vector_type(8)))  float  v8f;

static __device__ __forceinline__ v8f wmma_bf16(v16bf a, v16bf b, v8f c) {
  // D = A(16x32 bf16) * B(32x16 bf16) + C(16x16 f32)
  return __builtin_amdgcn_wmma_f32_16x16x32_bf16(false, a, false, b,
                                                 (short)0, c, false, false);
}

// A operand (16x32, bf16 source laid out [m][k], row stride ld).
// Per ISA: lanes 0-15 -> M=lane, K {0..7,16..23}; lanes 16-31 -> K {8..15,24..31}.
static __device__ __forceinline__ v16bf load_a_bf16(const bf16_t* src, int ld) {
  int lane = threadIdx.x & 31;
  int m = lane & 15, kh = lane >> 4;
  const bf16_t* p = src + (size_t)m * ld + kh * 8;
  v8bf lo = *(const v8bf*)(p);
  v8bf hi = *(const v8bf*)(p + 16);
  v16bf a;
#pragma unroll
  for (int t = 0; t < 8; ++t) { a[t] = lo[t]; a[8 + t] = hi[t]; }
  return a;
}

// Same A layout but fp32 source (weights), converted to bf16 in registers.
static __device__ __forceinline__ v16bf load_a_f32(const float* src, int ld) {
  int lane = threadIdx.x & 31;
  int m = lane & 15, kh = lane >> 4;
  const float* p = src + (size_t)m * ld + kh * 8;
  v16bf a;
#pragma unroll
  for (int t = 0; t < 8; ++t) { a[t] = (bf16_t)p[t]; a[8 + t] = (bf16_t)p[t + 16]; }
  return a;
}

// B operand (32x16). Storage is "n-major": srcT[n][k], k contiguous, row stride ld.
// Per ISA: lanes 0-15 hold K {0..7,16..23}, lanes 16-31 hold K {8..15,24..31}, N = lane&15.
static __device__ __forceinline__ v16bf load_b_bf16(const bf16_t* srcT, int ld, int k0) {
  int lane = threadIdx.x & 31;
  int n = lane & 15, kh = lane >> 4;
  const bf16_t* p = srcT + (size_t)n * ld + k0 + kh * 8;
  v8bf lo = *(const v8bf*)(p);
  v8bf hi = *(const v8bf*)(p + 16);
  v16bf b;
#pragma unroll
  for (int t = 0; t < 8; ++t) { b[t] = lo[t]; b[8 + t] = hi[t]; }
  return b;
}

struct ProjArgs { const float* W[6]; const float* bias[6]; };

// ---------------- Kernel 1: QKV projections (bf16 out) ----------------
// grid (32, 24): x = 128-wide n-block, y = proj*4 + batch. 256 thr = 8 waves.
// Q,K stored transposed [n][c] (packed b128 stores); V stored [c][n].
__global__ void k_proj(const float* __restrict__ x, ProjArgs pa,
                       bf16_t* __restrict__ qkv) {
  __shared__ bf16_t lX[128][72];   // [n][c], padded stride (144B rows)
  int nblk = blockIdx.x * 128;
  int pb = blockIdx.y;
  int p = pb >> 2, b = pb & 3;
  const float* xb = x + (size_t)b * CDIM * NDIM;
  int tid = threadIdx.x;
#pragma unroll
  for (int rep = 0; rep < 32; ++rep) {           // stage 64x128 fp32 -> bf16 LDS
    int flat = rep * 256 + tid;
    int c = flat >> 7, n = flat & 127;
    lX[n][c] = (bf16_t)xb[(size_t)c * NDIM + nblk + n];
  }
  __syncthreads();

  int w = tid >> 5, lane = tid & 31;
  int n0 = w * 16;
  v16bf b0 = load_b_bf16(&lX[n0][0], 72, 0);
  v16bf b1 = load_b_bf16(&lX[n0][0], 72, 32);

  const float* Wp = pa.W[p];
  const float* bp = pa.bias[p];
  bf16_t* dst = qkv + (size_t)pb * CDIM * NDIM;
  bool qk = (p % 3) != 2;          // uniform over the block
  int nl = lane & 15, hi = lane >> 4;
  int n = nblk + n0 + nl;
#pragma unroll
  for (int ot = 0; ot < 4; ++ot) {
    int o0 = ot * 16;
    v16bf a0 = load_a_f32(Wp + (size_t)o0 * CDIM, CDIM);
    v16bf a1 = load_a_f32(Wp + (size_t)o0 * CDIM + 32, CDIM);
    v8f acc = {};
    acc = wmma_bf16(a0, b0, acc);
    acc = wmma_bf16(a1, b1, acc);
    if (qk) {
      // [n][c]: lane's 8 rows are 8 consecutive channels -> one 16B store
      v8bf pk;
#pragma unroll
      for (int r = 0; r < 8; ++r)
        pk[r] = (bf16_t)(acc[r] + bp[o0 + 8 * hi + r]);
      *(v8bf*)(dst + (size_t)n * CDIM + o0 + 8 * hi) = pk;
    } else {
      // [c][n]: coalesced across the 16 lanes of each half
#pragma unroll
      for (int r = 0; r < 8; ++r) {
        int o = o0 + r + 8 * hi;
        dst[(size_t)o * NDIM + n] = (bf16_t)(acc[r] + bp[o]);
      }
    }
  }
}

// ---------------- Kernel 2: softmax row stats (m_i, l_i) ----------------
// grid (32, 8): x = block of 8 i-tiles (one/wave), y = branch*4 + batch.
__global__ void k_stats(const bf16_t* __restrict__ qkv,
                        float* __restrict__ mbuf, float* __restrict__ lbuf) {
  int bb = blockIdx.y;
  int branch = bb >> 2, b = bb & 3;
  int w = threadIdx.x >> 5, lane = threadIdx.x & 31;
  int i0 = (blockIdx.x * 8 + w) * 16;
  const bf16_t* qT = qkv + (size_t)((branch * 3 + 0) * NB + b) * CDIM * NDIM;
  const bf16_t* kT = qkv + (size_t)((branch * 3 + 1) * NB + b) * CDIM * NDIM;

  v16bf aq0 = load_a_bf16(qT + (size_t)i0 * CDIM, CDIM);
  v16bf aq1 = load_a_bf16(qT + (size_t)i0 * CDIM + 32, CDIM);

  float m_r[8], l_r[8];
#pragma unroll
  for (int r = 0; r < 8; ++r) { m_r[r] = -3.0e38f; l_r[r] = 0.f; }

  for (int j0 = 0; j0 < NDIM; j0 += 16) {
    __builtin_prefetch(kT + (size_t)(j0 + 16) * CDIM, 0, 1);  // global_prefetch_b8
    v16bf bk0 = load_b_bf16(kT + (size_t)j0 * CDIM, CDIM, 0);
    v16bf bk1 = load_b_bf16(kT + (size_t)j0 * CDIM, CDIM, 32);
    v8f s = {};
    s = wmma_bf16(aq0, bk0, s);
    s = wmma_bf16(aq1, bk1, s);
#pragma unroll
    for (int r = 0; r < 8; ++r) {
      float v = s[r];
      float cmax = v;
#pragma unroll
      for (int msk = 1; msk < 16; msk <<= 1)
        cmax = fmaxf(cmax, __shfl_xor(cmax, msk, 32));   // reduce over 16-lane row group
      float nm = fmaxf(m_r[r], cmax);
      float e = __expf(v - nm);
#pragma unroll
      for (int msk = 1; msk < 16; msk <<= 1)
        e += __shfl_xor(e, msk, 32);
      l_r[r] = l_r[r] * __expf(m_r[r] - nm) + e;
      m_r[r] = nm;
    }
  }
  if ((lane & 15) == 0) {
    int hi = lane >> 4;
#pragma unroll
    for (int r = 0; r < 8; ++r) {
      mbuf[(size_t)bb * NDIM + i0 + r + 8 * hi] = m_r[r];
      lbuf[(size_t)bb * NDIM + i0 + r + 8 * hi] = l_r[r];
    }
  }
}

// ---------------- Kernel 3: O = V * softmax(Q^T K), gated, -> catT ----------------
// grid (32, 8): x = block of 8 j-tiles (one/wave), y = branch*4 + batch.
__global__ void k_out(const bf16_t* __restrict__ qkv,
                      const float* __restrict__ mbuf, const float* __restrict__ lbuf,
                      const float* __restrict__ g1, const float* __restrict__ g2,
                      bf16_t* __restrict__ catT) {
  __shared__ bf16_t pbuf[8][16][40];   // per-wave P tile as [j][i_local], 80B rows
  int bb = blockIdx.y;
  int branch = bb >> 2, b = bb & 3;
  int w = threadIdx.x >> 5, lane = threadIdx.x & 31;
  int j0 = (blockIdx.x * 8 + w) * 16;
  const bf16_t* qT = qkv + (size_t)((branch * 3 + 0) * NB + b) * CDIM * NDIM;
  const bf16_t* kT = qkv + (size_t)((branch * 3 + 1) * NB + b) * CDIM * NDIM;
  const bf16_t* V  = qkv + (size_t)((branch * 3 + 2) * NB + b) * CDIM * NDIM;
  const float* mrow = mbuf + (size_t)bb * NDIM;
  const float* lrow = lbuf + (size_t)bb * NDIM;

  v16bf bkj0 = load_b_bf16(kT + (size_t)j0 * CDIM, CDIM, 0);   // K tile constant over i
  v16bf bkj1 = load_b_bf16(kT + (size_t)j0 * CDIM, CDIM, 32);

  v8f acc[4] = {{}, {}, {}, {}};
  int nl = lane & 15, hi = lane >> 4;

  for (int i0 = 0; i0 < NDIM; i0 += 32) {
    __builtin_prefetch(qT + (size_t)(i0 + 32) * CDIM, 0, 1);
    __builtin_prefetch(V + i0 + 32, 0, 1);
#pragma unroll
    for (int it = 0; it < 2; ++it) {              // two 16x16 score tiles
      int ib = i0 + it * 16;
      v16bf aq0 = load_a_bf16(qT + (size_t)ib * CDIM, CDIM);
      v16bf aq1 = load_a_bf16(qT + (size_t)ib * CDIM + 32, CDIM);
      v8f s = {};
      s = wmma_bf16(aq0, bkj0, s);
      s = wmma_bf16(aq1, bkj1, s);
#pragma unroll
      for (int r = 0; r < 8; ++r) {
        int i = ib + r + 8 * hi;
        float pv = __expf(s[r] - mrow[i]) * (1.0f / lrow[i]);
        pbuf[w][nl][it * 16 + r + 8 * hi] = (bf16_t)pv;   // transpose via LDS
      }
    }
    // same-wave LDS RAW: DS ops are in-order; compiler inserts s_wait_dscnt
    v16bf bp = load_b_bf16(&pbuf[w][0][0], 40, 0);        // K=32 over i_local
#pragma unroll
    for (int ct = 0; ct < 4; ++ct) {
      v16bf av = load_a_bf16(V + (size_t)(ct * 16) * NDIM + i0, NDIM); // A[m=c][k=i]
      acc[ct] = wmma_bf16(av, bp, acc[ct]);
    }
  }

  float g = (branch == 0) ? g1[0] : g2[0];
  bf16_t* dst = catT + ((size_t)b * NDIM + j0 + nl) * (2 * CDIM) + branch * CDIM;
#pragma unroll
  for (int ct = 0; ct < 4; ++ct) {
    v8bf pk;
#pragma unroll
    for (int r = 0; r < 8; ++r) pk[r] = (bf16_t)(acc[ct][r] * g);
    *(v8bf*)(dst + ct * 16 + 8 * hi) = pk;     // 8 consecutive channels, 16B store
  }
}

// ---------------- Kernel 4: fused 1x1 conv (K=128) + bias -> fp32 out ----------------
// grid (32, 4): x = 128-wide n-block (8 waves x 16), y = batch.
__global__ void k_fuse(const bf16_t* __restrict__ catT, const float* __restrict__ Wf,
                       const float* __restrict__ bfv, float* __restrict__ out) {
  int b = blockIdx.y;
  int w = threadIdx.x >> 5, lane = threadIdx.x & 31;
  int n0 = blockIdx.x * 128 + w * 16;
  const bf16_t* src = catT + ((size_t)b * NDIM + n0) * 128;
  v16bf bc[4];
#pragma unroll
  for (int kc = 0; kc < 4; ++kc) bc[kc] = load_b_bf16(src, 128, kc * 32);
  int nl = lane & 15, hi = lane >> 4;
#pragma unroll
  for (int ot = 0; ot < 4; ++ot) {
    int o0 = ot * 16;
    v8f acc = {};
#pragma unroll
    for (int kc = 0; kc < 4; ++kc) {
      v16bf a = load_a_f32(Wf + (size_t)o0 * 128 + kc * 32, 128);
      acc = wmma_bf16(a, bc[kc], acc);
    }
#pragma unroll
    for (int r = 0; r < 8; ++r) {
      int o = o0 + r + 8 * hi;
      out[((size_t)b * CDIM + o) * NDIM + n0 + nl] = acc[r] + bfv[o];
    }
  }
}

extern "C" void kernel_launch(void* const* d_in, const int* in_sizes, int n_in,
                              void* d_out, int out_size, void* d_ws, size_t ws_size,
                              hipStream_t stream) {
  const float* x = (const float*)d_in[0];
  ProjArgs pa;
  for (int p = 0; p < 6; ++p) {
    pa.W[p]    = (const float*)d_in[1 + 2 * p];
    pa.bias[p] = (const float*)d_in[2 + 2 * p];
  }
  const float* g1  = (const float*)d_in[13];
  const float* g2  = (const float*)d_in[14];
  const float* Wf  = (const float*)d_in[15];
  const float* bfv = (const float*)d_in[16];

  // Workspace layout (bytes):
  //   qkv   : 6 proj * 4 batch * 64*4096 bf16 = 12,582,912
  //   mbuf  : 8 * 4096 f32                    =    131,072
  //   lbuf  : 8 * 4096 f32                    =    131,072
  //   catT  : 4 * 4096 * 128 bf16             =  4,194,304
  char* ws = (char*)d_ws;
  bf16_t* qkv  = (bf16_t*)ws;
  float*  mbuf = (float*)(ws + 12582912);
  float*  lbuf = (float*)(ws + 12582912 + 131072);
  bf16_t* catT = (bf16_t*)(ws + 12582912 + 262144);
  float*  out  = (float*)d_out;

  k_proj <<<dim3(32, 24), 256, 0, stream>>>(x, pa, qkv);
  k_stats<<<dim3(32, 8),  256, 0, stream>>>(qkv, mbuf, lbuf);
  k_out  <<<dim3(32, 8),  256, 0, stream>>>(qkv, mbuf, lbuf, g1, g2, catT);
  k_fuse <<<dim3(32, 4),  256, 0, stream>>>(catT, Wf, bfv, out);
}
